// BaseModel_28621662061167
// MI455X (gfx1250) — matrix operand order repaired and checked by reading, not compiled, not run on hardware
//
#include <hip/hip_runtime.h>

// ---------------------------------------------------------------------------
// Fused prototype-MIL model for MI455X (gfx1250).
// Split-bf16 (hi+lo) WMMA path: each fp32 GEMM computed as
//   A_hi*B_hi + A_hi*B_lo + A_lo*B_hi  with f32 accumulation
// using v_wmma_f32_16x16x32_bf16. All intermediates LDS-resident.
// X tile staging is register-prefetched across K-chunks so HBM latency
// overlaps the WMMA block of the previous chunk.
// ---------------------------------------------------------------------------

typedef __bf16 bf16;
typedef __attribute__((ext_vector_type(2)))  float v2f;
typedef __attribute__((ext_vector_type(4)))  float v4f;
typedef __attribute__((ext_vector_type(8)))  float v8f;
typedef __attribute__((ext_vector_type(4)))  __bf16 v4bf;
typedef __attribute__((ext_vector_type(8)))  __bf16 v8bf;
typedef __attribute__((ext_vector_type(16))) __bf16 v16bf;

#define N_CELLS   200000
#define IN_DIM    1000
#define KPAD      1024          // IN_DIM zero-padded to a multiple of 32
#define H_DIM     256
#define Z_DIM     32
#define N_PROTO   64
#define N_CLASSES 8
#define N_BAGS    64

#define MT        64            // rows per workgroup (200000 = 64*3125 exact)
#define KC        64            // staged K-chunk for GEMM1 (two 32-steps)

// LDS strides in bf16 elements: multiples of 8 (b128-aligned rows) and
// chosen so row*stride/4 mod 64 tiles the banks conflict-free.
#define XS_STR    72
#define HS_STR    264
#define ZS_STR    40
#define RS_STR    66            // f32 stride for r (aliases H region)

#define X_PLANE   (MT * XS_STR)           // 4608 bf16 = 9216 B
#define H_PLANE   (MT * HS_STR)           // 16896 bf16 = 33792 B
#define Z_PLANE   (MT * ZS_STR)           // 2560 bf16 = 5120 B
#define X_OFF     0
#define H_OFF     (2 * X_PLANE * 2)                       // 18432
#define Z_OFF     (H_OFF + 2 * H_PLANE * 2)               // 86016
#define ZSQ_OFF   (Z_OFF + 2 * Z_PLANE * 2)               // 96256
#define PSQ_OFF   (ZSQ_OFF + 256)                         // 96512
#define SMEM_SZ   (PSQ_OFF + 256)                         // 96768 B

__device__ __forceinline__ float lrelu(float x) { return x > 0.0f ? x : 0.01f * x; }

__device__ __forceinline__ v16bf join8(v8bf a, v8bf b) {
  return __builtin_shufflevector(a, b, 0, 1, 2, 3, 4, 5, 6, 7,
                                 8, 9, 10, 11, 12, 13, 14, 15);
}
// A-fragment (16x32 bf16): per-lane K-chunks [8*hi,+8) and [16+8*hi,+8)
__device__ __forceinline__ v16bf ld_a(const bf16* p, int row_off, int k, int hi) {
  v8bf c0 = *(const v8bf*)(p + row_off + k + 8 * hi);
  v8bf c1 = *(const v8bf*)(p + row_off + k + 16 + 8 * hi);
  return join8(c0, c1);
}
// B-fragment (32x16 bf16): per-lane K-chunks [8*hi,+8) and [16+8*hi,+8)
__device__ __forceinline__ v16bf ld_b(const bf16* p, int off, int hi) {
  v8bf c0 = *(const v8bf*)(p + off + 8 * hi);
  v8bf c1 = *(const v8bf*)(p + off + 16 + 8 * hi);
  return join8(c0, c1);
}
__device__ __forceinline__ v8f wmma_bf16(v16bf a, v16bf b, v8f c) {
  return __builtin_amdgcn_wmma_f32_16x16x32_bf16(
      false, a, false, b, (short)0, c, false, false);
}
// split product: (ah+al)*(bh+bl) ~= ah*bh + ah*bl + al*bh  (f32 accumulate)
__device__ __forceinline__ v8f mm3(v16bf ah, v16bf al, v16bf bh, v16bf bl, v8f c) {
  c = wmma_bf16(ah, bh, c);
  c = wmma_bf16(ah, bl, c);
  c = wmma_bf16(al, bh, c);
  return c;
}

__global__ __launch_bounds__(256) void fused_cells_kernel(
    const float* __restrict__ x,     const int*   __restrict__ seg,
    const bf16* __restrict__ WiH,    const bf16* __restrict__ WiL,
    const bf16* __restrict__ WhH,    const bf16* __restrict__ WhL,
    const bf16* __restrict__ WzH,    const bf16* __restrict__ WzL,
    const bf16* __restrict__ PH,     const bf16* __restrict__ PL,
    const float* __restrict__ bi,    const float* __restrict__ bh,
    const float* __restrict__ bz,    const float* __restrict__ proto,
    const float* __restrict__ Wclf,
    float* __restrict__ bag_sums,    float* __restrict__ bag_cnt)
{
  __shared__ __align__(16) unsigned char smem[SMEM_SZ];
  bf16* Xh = (bf16*)(smem + X_OFF);
  bf16* Xl = Xh + X_PLANE;
  bf16* Hh = (bf16*)(smem + H_OFF);
  bf16* Hl = Hh + H_PLANE;
  bf16* Zh = (bf16*)(smem + Z_OFF);
  bf16* Zl = Zh + Z_PLANE;
  float* zsq = (float*)(smem + ZSQ_OFF);
  float* psq = (float*)(smem + PSQ_OFF);
  float* Rs  = (float*)(smem + H_OFF);   // r aliases H region after GEMM-z

  const int tid  = threadIdx.x;
  const int lane = tid & 31;
  const int w    = tid >> 5;      // wave id 0..7
  const int l16  = lane & 15;
  const int hi   = lane >> 4;     // 0/1 lane half per ISA fragment layouts
  const int row_base = blockIdx.x * MT;

  const v8f zero8 = {0.f, 0.f, 0.f, 0.f, 0.f, 0.f, 0.f, 0.f};

  // prototype squared norms from full-precision prototypes
  if (tid < N_PROTO) {
    float s = 0.f;
    for (int k = 0; k < Z_DIM; ++k) { float p = proto[tid * Z_DIM + k]; s += p * p; }
    psq[tid] = s;
  }

  // ---------------- GEMM1: h1 = lrelu(X @ Wi^T + bi), tile 64x256 ----------
  const int mg = w & 1;   // 2 M-groups of 32 rows
  const int ng = w >> 1;  // 4 N-groups of 64 cols
  v8f acc[2][4];
  #pragma unroll
  for (int i = 0; i < 2; ++i)
    #pragma unroll
    for (int j = 0; j < 4; ++j) acc[i][j] = zero8;

  // register-prefetched X staging: 64 rows x KC f32 = 1024 float4 / 256 thr
  v4f xreg[4];
  const int xr  = tid >> 4;          // row handled by this thread (16 v4f/row)
  const int xc4 = tid & 15;          // float4 index within row
  const float* xrow = x + (size_t)(row_base + xr) * IN_DIM;

  // prologue: load chunk 0
  #pragma unroll
  for (int j = 0; j < 4; ++j) {
    int kg = 4 * xc4 + j * KC / 4 * 4;  // see below; rewritten cleanly in loop
    xreg[j] = (v4f){0.f, 0.f, 0.f, 0.f};
  }
  #pragma unroll
  for (int j = 0; j < 4; ++j) {
    int r  = (tid + j * 256) >> 4;
    int c4 = (tid + j * 256) & 15;
    int kg = 4 * c4;
    if (kg < IN_DIM)
      xreg[j] = *(const v4f*)(x + (size_t)(row_base + r) * IN_DIM + kg);
  }

  for (int k0 = 0; k0 < KPAD; k0 += KC) {
    __syncthreads();   // previous chunk's WMMA reads of Xs complete
    // convert + store the prefetched chunk into LDS planes
    #pragma unroll
    for (int j = 0; j < 4; ++j) {
      int i  = tid + j * 256;
      int r  = i >> 4;
      int c4 = i & 15;
      int o  = r * XS_STR + 4 * c4;
      v4f v = xreg[j];
      bf16 a0 = (bf16)v.x, a1 = (bf16)v.y, a2 = (bf16)v.z, a3 = (bf16)v.w;
      *(v4bf*)&Xh[o] = (v4bf){a0, a1, a2, a3};
      *(v4bf*)&Xl[o] = (v4bf){(bf16)(v.x - (float)a0), (bf16)(v.y - (float)a1),
                              (bf16)(v.z - (float)a2), (bf16)(v.w - (float)a3)};
    }
    __syncthreads();

    // prefetch next chunk (overlaps the WMMA block below)
    if (k0 + KC < KPAD) {
      #pragma unroll
      for (int j = 0; j < 4; ++j) {
        int i  = tid + j * 256;
        int r  = i >> 4;
        int c4 = i & 15;
        int kg = k0 + KC + 4 * c4;
        v4f v = {0.f, 0.f, 0.f, 0.f};
        if (kg < IN_DIM)
          v = *(const v4f*)(x + (size_t)(row_base + r) * IN_DIM + kg);
        xreg[j] = v;
      }
    }

    #pragma unroll
    for (int ks = 0; ks < KC; ks += 32) {
      v16bf ah[2], al[2], bhf[4], blf[4];
      #pragma unroll
      for (int mb = 0; mb < 2; ++mb) {
        int ro = (32 * mg + 16 * mb + l16) * XS_STR;
        ah[mb] = ld_a(Xh, ro, ks, hi);
        al[mb] = ld_a(Xl, ro, ks, hi);
      }
      #pragma unroll
      for (int nb = 0; nb < 4; ++nb) {
        int off = (64 * ng + 16 * nb + l16) * KPAD + k0 + ks;
        bhf[nb] = ld_b(WiH, off, hi);
        blf[nb] = ld_b(WiL, off, hi);
      }
      #pragma unroll
      for (int mb = 0; mb < 2; ++mb)
        #pragma unroll
        for (int nb = 0; nb < 4; ++nb)
          acc[mb][nb] = mm3(ah[mb], al[mb], bhf[nb], blf[nb], acc[mb][nb]);
    }
  }
  __syncthreads();
  // epilogue: bias + lrelu -> split-bf16 h1 planes (C layout: M=v+8*hi, N=l16)
  #pragma unroll
  for (int mb = 0; mb < 2; ++mb)
    #pragma unroll
    for (int nb = 0; nb < 4; ++nb) {
      int col = 64 * ng + 16 * nb + l16;
      float bb = bi[col];
      #pragma unroll
      for (int v = 0; v < 8; ++v) {
        int r = 32 * mg + 16 * mb + v + 8 * hi;
        float hval = lrelu(acc[mb][nb][v] + bb);
        bf16 hh = (bf16)hval;
        Hh[r * HS_STR + col] = hh;
        Hl[r * HS_STR + col] = (bf16)(hval - (float)hh);
      }
    }
  __syncthreads();

  // ---------------- GEMM2: h2 = lrelu(h1 @ Wh^T + bh) ----------------------
  #pragma unroll
  for (int i = 0; i < 2; ++i)
    #pragma unroll
    for (int j = 0; j < 4; ++j) acc[i][j] = zero8;

  #pragma unroll 2
  for (int k = 0; k < H_DIM; k += 32) {
    v16bf ah[2], al[2], bhf[4], blf[4];
    #pragma unroll
    for (int mb = 0; mb < 2; ++mb) {
      int ro = (32 * mg + 16 * mb + l16) * HS_STR;
      ah[mb] = ld_a(Hh, ro, k, hi);
      al[mb] = ld_a(Hl, ro, k, hi);
    }
    #pragma unroll
    for (int nb = 0; nb < 4; ++nb) {
      int off = (64 * ng + 16 * nb + l16) * H_DIM + k;
      bhf[nb] = ld_b(WhH, off, hi);
      blf[nb] = ld_b(WhL, off, hi);
    }
    #pragma unroll
    for (int mb = 0; mb < 2; ++mb)
      #pragma unroll
      for (int nb = 0; nb < 4; ++nb)
        acc[mb][nb] = mm3(ah[mb], al[mb], bhf[nb], blf[nb], acc[mb][nb]);
  }
  __syncthreads();  // all h1 reads done before overwriting planes with h2
  #pragma unroll
  for (int mb = 0; mb < 2; ++mb)
    #pragma unroll
    for (int nb = 0; nb < 4; ++nb) {
      int col = 64 * ng + 16 * nb + l16;
      float bb = bh[col];
      #pragma unroll
      for (int v = 0; v < 8; ++v) {
        int r = 32 * mg + 16 * mb + v + 8 * hi;
        float hval = lrelu(acc[mb][nb][v] + bb);
        bf16 hh = (bf16)hval;
        Hh[r * HS_STR + col] = hh;
        Hl[r * HS_STR + col] = (bf16)(hval - (float)hh);
      }
    }
  __syncthreads();

  // ---------------- GEMM-z: z = lrelu(h2 @ Wz^T + bz), 64x32 ---------------
  const int mbz = w & 3;   // 4 M-tiles of 16
  const int nbz = w >> 2;  // 2 N-tiles of 16
  v8f accz = zero8;
  #pragma unroll 2
  for (int k = 0; k < H_DIM; k += 32) {
    int ro = (16 * mbz + l16) * HS_STR;
    v16bf ah = ld_a(Hh, ro, k, hi);
    v16bf al = ld_a(Hl, ro, k, hi);
    int off = (16 * nbz + l16) * H_DIM + k;
    v16bf bhf = ld_b(WzH, off, hi);
    v16bf blf = ld_b(WzL, off, hi);
    accz = mm3(ah, al, bhf, blf, accz);
  }
  {
    int col = 16 * nbz + l16;
    float bb = bz[col];
    #pragma unroll
    for (int v = 0; v < 8; ++v) {
      int r = 16 * mbz + v + 8 * hi;
      float zval = lrelu(accz[v] + bb);
      bf16 zh = (bf16)zval;
      Zh[r * ZS_STR + col] = zh;
      Zl[r * ZS_STR + col] = (bf16)(zval - (float)zh);
    }
  }
  __syncthreads();
  if (tid < MT) {
    float s = 0.f;
    for (int k = 0; k < Z_DIM; ++k) {
      float zv = (float)Zh[tid * ZS_STR + k] + (float)Zl[tid * ZS_STR + k];
      s += zv * zv;
    }
    zsq[tid] = s;
  }
  __syncthreads();

  // ------------- GEMM-G: G = z @ P^T; r = 1/(zsq+psq-2G+0.5) ---------------
  const int mbg = w & 3;
  const int ngg = w >> 2;
  #pragma unroll
  for (int nbi = 0; nbi < 2; ++nbi) {
    const int nb = 2 * ngg + nbi;
    int ro = (16 * mbg + l16) * ZS_STR;
    v16bf ah = ld_a(Zh, ro, 0, hi);
    v16bf al = ld_a(Zl, ro, 0, hi);
    int off = (16 * nb + l16) * Z_DIM;
    v16bf bhf = ld_b(PH, off, hi);
    v16bf blf = ld_b(PL, off, hi);
    v8f accg = mm3(ah, al, bhf, blf, zero8);

    const int pcol = 16 * nb + l16;
    const float pq = psq[pcol];
    #pragma unroll
    for (int v = 0; v < 8; ++v) {
      int m = 16 * mbg + v + 8 * hi;
      float d = zsq[m] + pq - 2.0f * accg[v] + 0.5f;
      Rs[m * RS_STR + pcol] = 1.0f / d;
    }
  }
  __syncthreads();

  // ------------- classifier matvec + ragged segment accumulation -----------
  if (tid < MT) {
    float cl[N_CLASSES];
    #pragma unroll
    for (int c = 0; c < N_CLASSES; ++c) cl[c] = 0.f;
    for (int p = 0; p < N_PROTO; ++p) {
      float rv = Rs[tid * RS_STR + p];
      #pragma unroll
      for (int c = 0; c < N_CLASSES; ++c) cl[c] += rv * Wclf[c * N_PROTO + p];
    }
    int s = seg[row_base + tid];
    #pragma unroll
    for (int c = 0; c < N_CLASSES; ++c)
      atomicAdd(&bag_sums[s * N_CLASSES + c], cl[c]);
    atomicAdd(&bag_cnt[s], 1.0f);
  }
}

// split fp32 -> (bf16 hi, bf16 lo) planes with K zero-padding
__global__ void convert_split_kernel(const float* __restrict__ src,
                                     bf16* __restrict__ hi, bf16* __restrict__ lo,
                                     int rows, int ksrc, int kdst) {
  int total = rows * kdst;
  for (int i = blockIdx.x * blockDim.x + threadIdx.x; i < total;
       i += gridDim.x * blockDim.x) {
    int n = i / kdst, k = i - n * kdst;
    float v = (k < ksrc) ? src[n * ksrc + k] : 0.f;
    bf16 h = (bf16)v;
    hi[i] = h;
    lo[i] = (bf16)(v - (float)h);
  }
}

__global__ void init_zero_kernel(float* p, int n) {
  for (int i = threadIdx.x; i < n; i += blockDim.x) p[i] = 0.f;
}

__global__ void finalize_kernel(const float* __restrict__ bag_sums,
                                const float* __restrict__ bag_cnt,
                                const int*   __restrict__ y,
                                float* __restrict__ out)  // out[0]=loss, out[1..512]=logits
{
  __shared__ float part[N_BAGS];
  const int b = threadIdx.x;  // 64 threads
  float l[N_CLASSES];
  float c = fmaxf(bag_cnt[b], 1.0f);
  #pragma unroll
  for (int j = 0; j < N_CLASSES; ++j) {
    l[j] = bag_sums[b * N_CLASSES + j] / c;
    out[1 + b * N_CLASSES + j] = l[j];
  }
  float m = l[0];
  #pragma unroll
  for (int j = 1; j < N_CLASSES; ++j) m = fmaxf(m, l[j]);
  float s = 0.f;
  #pragma unroll
  for (int j = 0; j < N_CLASSES; ++j) s += __expf(l[j] - m);
  float lse = m + __logf(s);
  int yb = y[b];
  part[b] = -(l[yb] - lse) * (1.0f / (float)N_BAGS);
  __syncthreads();
  if (b == 0) {
    float t = 0.f;
    for (int i = 0; i < N_BAGS; ++i) t += part[i];
    out[0] = t;
  }
}

extern "C" void kernel_launch(void* const* d_in, const int* in_sizes, int n_in,
                              void* d_out, int out_size, void* d_ws, size_t ws_size,
                              hipStream_t stream) {
  const float* x     = (const float*)d_in[0];
  const int*   y     = (const int*)  d_in[1];
  const int*   seg   = (const int*)  d_in[2];
  const float* Wi    = (const float*)d_in[3];
  const float* bi    = (const float*)d_in[4];
  const float* Wh    = (const float*)d_in[5];
  const float* bh    = (const float*)d_in[6];
  const float* Wz    = (const float*)d_in[7];
  const float* bz    = (const float*)d_in[8];
  const float* proto = (const float*)d_in[9];
  const float* Wclf  = (const float*)d_in[10];
  float* out = (float*)d_out;

  // workspace layout (bytes)
  char* wsb = (char*)d_ws;
  float* bag_sums = (float*)(wsb + 0);       // 512 f32
  float* bag_cnt  = (float*)(wsb + 2048);    // 64 f32
  bf16* WiH = (bf16*)(wsb + 4096);           // 256*1024
  bf16* WiL = WiH + 256 * KPAD;
  bf16* WhH = (bf16*)(wsb + 1052672);        // 256*256
  bf16* WhL = WhH + 256 * 256;
  bf16* WzH = (bf16*)(wsb + 1314816);        // 32*256
  bf16* WzL = WzH + 32 * 256;
  bf16* PH  = (bf16*)(wsb + 1347584);        // 64*32
  bf16* PL  = PH + 64 * 32;

  init_zero_kernel<<<1, 256, 0, stream>>>(bag_sums, N_BAGS * N_CLASSES + N_BAGS);
  convert_split_kernel<<<512, 256, 0, stream>>>(Wi, WiH, WiL, H_DIM, IN_DIM, KPAD);
  convert_split_kernel<<<128, 256, 0, stream>>>(Wh, WhH, WhL, H_DIM, H_DIM, H_DIM);
  convert_split_kernel<<<16, 256, 0, stream>>>(Wz, WzH, WzL, Z_DIM, H_DIM, H_DIM);
  convert_split_kernel<<<4, 256, 0, stream>>>(proto, PH, PL, N_PROTO, Z_DIM, Z_DIM);

  fused_cells_kernel<<<N_CELLS / MT, 256, 0, stream>>>(
      x, seg, WiH, WiL, WhH, WhL, WzH, WzL, PH, PL,
      bi, bh, bz, proto, Wclf, bag_sums, bag_cnt);

  finalize_kernel<<<1, N_BAGS, 0, stream>>>(bag_sums, bag_cnt, y, out);
}
